// MultiheadAttention2_28991029248212
// MI455X (gfx1250) — matrix operand order repaired and checked by reading, not compiled, not run on hardware
//
#include <hip/hip_runtime.h>

typedef _Float16 half_t;
typedef __attribute__((ext_vector_type(16))) _Float16 v16h;
typedef __attribute__((ext_vector_type(8)))  _Float16 v8h;
typedef __attribute__((ext_vector_type(8)))  float    v8f;

#define T_LEN   1024
#define BATCH   4
#define EMB     1024
#define HEADS   16
#define HDIM    64
#define ROWS    4096      // T*B
#define THREE_E 3072

__device__ __forceinline__ v8f wmma16(v16h a, v16h b, v8f c) {
  // D = A(16x32 f16) x B(32x16 f16) + C(16x16 f32)
  return __builtin_amdgcn_wmma_f32_16x16x32_f16(false, a, false, b, (short)0, c,
                                                false, false);
}

// 16x32 f16 A-tile chunk from row-major memory, ISA A layout:
// lane m = lane&15 (M); per-lane slot j holds K = k0 + j + 8*(j>=8) + 8*(lane>=16)
__device__ __forceinline__ v16h load_a(const half_t* __restrict__ base, int ld,
                                       int lane, int k0) {
  const v8h* p = (const v8h*)(base + (size_t)(lane & 15) * ld + k0 +
                              ((lane >> 4) << 3));
  v8h lo = p[0];
  v8h hi = p[2];                   // +16 halfs
  v16h a;
#pragma unroll
  for (int j = 0; j < 8; ++j) { a[j] = lo[j]; a[j + 8] = hi[j]; }
  return a;
}

// B tile: lane n = lane&15 (N); slot j holds K = j + 8*(j>=8) + 8*(lane>=16)
__device__ __forceinline__ v16h load_b(const half_t* __restrict__ p, int ld) {
  int lane = threadIdx.x & 31;
  const half_t* q = p - (size_t)lane * ld + (lane & 15) + (size_t)((lane >> 4) << 3) * ld;
  v16h b;
#pragma unroll
  for (int j = 0; j < 8; ++j) {
    b[j] = q[(size_t)j * ld];
    b[j + 8] = q[(size_t)(16 + j) * ld];
  }
  return b;
}

// A tile from f32 LDS (normalized P) with cvt to f16
__device__ __forceinline__ v16h pack_p(const float* __restrict__ Sld, int lane,
                                       int kc) {
  const float* prow =
      Sld + (size_t)(lane & 15) * 1024 + kc + ((lane >> 4) << 3);
  v16h ap;
#pragma unroll
  for (int j = 0; j < 8; ++j) {
    ap[j] = (half_t)prow[j];
    ap[j + 8] = (half_t)prow[16 + j];
  }
  return ap;
}

// ---------------------------------------------------------------- conversions
__global__ void cvt_f32_to_f16(const float* __restrict__ in,
                               half_t* __restrict__ out, int n) {
  int i = blockIdx.x * blockDim.x + threadIdx.x;
  int stride = gridDim.x * blockDim.x;
  for (; i < n; i += stride) out[i] = (half_t)in[i];
}

// out[c*R + r] = (f16) in[r*C + c]   (LDS-tiled, coalesced both sides)
__global__ void __launch_bounds__(256) transpose_f32_f16(
    const float* __restrict__ in, half_t* __restrict__ out, int R, int C) {
  __shared__ float tile[32][33];
  int ctiles = C >> 5;
  int c0 = (blockIdx.x % ctiles) << 5;
  int r0 = (blockIdx.x / ctiles) << 5;
  int tx = threadIdx.x & 31, ty = threadIdx.x >> 5;   // 32 x 8
#pragma unroll
  for (int i = 0; i < 32; i += 8)
    tile[ty + i][tx] = in[(size_t)(r0 + ty + i) * C + c0 + tx];
  __syncthreads();
#pragma unroll
  for (int i = 0; i < 32; i += 8)
    out[(size_t)(c0 + ty + i) * R + r0 + tx] = (half_t)tile[tx][ty + i];
}

// ctx[t*B+b] = dot(query[t,b,:], context_w)
__global__ void ctx_kernel(const float* __restrict__ q,
                           const float* __restrict__ cw,
                           float* __restrict__ ctx) {
  __shared__ float red[256];
  int r = blockIdx.x;
  float s = 0.f;
  for (int e = threadIdx.x; e < EMB; e += 256)
    s += q[(size_t)r * EMB + e] * cw[e];
  red[threadIdx.x] = s;
  __syncthreads();
  for (int off = 128; off > 0; off >>= 1) {
    if ((int)threadIdx.x < off) red[threadIdx.x] += red[threadIdx.x + off];
    __syncthreads();
  }
  if (threadIdx.x == 0) ctx[r] = red[0];
}

// ------------------------------------------------------------------- QKV GEMM
// 16x64 output tile per wave, double-buffered K loop; whT is (E x 3E).
// Scatter: q -> (B,H,T,D) scaled, k -> (B,H,D,T) TRANSPOSED, v -> (B,H,T,D)
__global__ void __launch_bounds__(256) qkv_kernel(
    const half_t* __restrict__ qh, const half_t* __restrict__ whT,
    const float* __restrict__ bias, half_t* __restrict__ qb,
    half_t* __restrict__ kTb, half_t* __restrict__ vb) {
  int lane = threadIdx.x & 31;
  int wave = threadIdx.x >> 5;
  int gw = blockIdx.x * 8 + wave;
  int tn = gw % (THREE_E / 64);        // 48 column groups of 64
  int tm = gw / (THREE_E / 64);
  int tb0 = tm * 16, f0 = tn * 64;
  v8f acc[4] = {};
  const half_t* abase = qh + (size_t)tb0 * EMB;

  v16h a = load_a(abase, EMB, lane, 0);
  const half_t* brow = whT + (size_t)lane * THREE_E + f0;
  v16h b0 = load_b(brow, THREE_E), b1 = load_b(brow + 16, THREE_E);
  v16h b2 = load_b(brow + 32, THREE_E), b3 = load_b(brow + 48, THREE_E);
  for (int k0 = 0; k0 < EMB - 32; k0 += 32) {
    const half_t* brn = whT + (size_t)(k0 + 32 + lane) * THREE_E + f0;
    v16h an = load_a(abase, EMB, lane, k0 + 32);
    v16h n0 = load_b(brn, THREE_E), n1 = load_b(brn + 16, THREE_E);
    v16h n2 = load_b(brn + 32, THREE_E), n3 = load_b(brn + 48, THREE_E);
    __builtin_prefetch(abase + (size_t)(lane & 15) * EMB + k0 + 64, 0, 1);
    __builtin_prefetch(brn + (size_t)32 * THREE_E, 0, 1);
    acc[0] = wmma16(a, b0, acc[0]);
    acc[1] = wmma16(a, b1, acc[1]);
    acc[2] = wmma16(a, b2, acc[2]);
    acc[3] = wmma16(a, b3, acc[3]);
    a = an; b0 = n0; b1 = n1; b2 = n2; b3 = n3;
  }
  acc[0] = wmma16(a, b0, acc[0]);
  acc[1] = wmma16(a, b1, acc[1]);
  acc[2] = wmma16(a, b2, acc[2]);
  acc[3] = wmma16(a, b3, acc[3]);

  int n = lane & 15, m_add = (lane >> 4) << 3;
#pragma unroll
  for (int jt = 0; jt < 4; ++jt) {
    int f = f0 + jt * 16 + n;
    float bfv = bias[f];
#pragma unroll
    for (int r = 0; r < 8; ++r) {
      int tb = tb0 + r + m_add;
      int t = tb >> 2, bb = tb & 3;               // rows are t*B+b, B=4
      float val = acc[jt][r] + bfv;
      if (f < 1024) {
        int hh = f >> 6, d = f & 63;
        qb[(((size_t)(bb * HEADS + hh)) * T_LEN + t) * HDIM + d] =
            (half_t)(val * 0.125f);               // q * D^-0.5
      } else if (f < 2048) {
        int f2 = f - 1024; int hh = f2 >> 6, d = f2 & 63;
        kTb[(((size_t)(bb * HEADS + hh)) * HDIM + d) * T_LEN + t] = (half_t)val;
      } else {
        int f2 = f - 2048; int hh = f2 >> 6, d = f2 & 63;
        vb[(((size_t)(bb * HEADS + hh)) * T_LEN + t) * HDIM + d] = (half_t)val;
      }
    }
  }
}

// ----------------------------------------------------------- fused attention
// One block per (b, 16-row tile); loops heads. QK^T + bias -> softmax with P
// materialized in LDS (needed for avg_weights) -> P@V -> relu -> f16 store.
__global__ void __launch_bounds__(256) attn_kernel(
    const half_t* __restrict__ qb, const half_t* __restrict__ kTb,
    const half_t* __restrict__ vb, const float* __restrict__ ctx,
    const float* __restrict__ fx, const float* __restrict__ bx,
    const float* __restrict__ hx, half_t* __restrict__ arelu,
    float* __restrict__ avg_out) {
  extern __shared__ float smem[];
  float* Sld = smem;               // 16*1024 f32 score/P row-block (64 KB)
  float* red = smem + 16384;       // 256 f32 reduction scratch
  int tid = threadIdx.x;
  int lane = tid & 31, wave = tid >> 5;
  int b = blockIdx.x >> 6;                      // T/16 = 64 tiles per batch
  int t0 = (blockIdx.x & 63) << 4;
  int row = tid >> 4, sub = tid & 15;

  for (int h = 0; h < HEADS; ++h) {
    size_t hb = ((size_t)(b * HEADS + h)) * T_LEN;
    const half_t* qbase = qb + (hb + t0) * HDIM;
    const half_t* kTh = kTb + ((size_t)(b * HEADS + h)) * HDIM * T_LEN;
    v16h a0 = load_a(qbase, HDIM, lane, 0);
    v16h a1 = load_a(qbase, HDIM, lane, 32);
    float hxv = hx[h];

    // ---- scores: wave covers key cols [wave*128, +128), pipelined K tiles
    const half_t* kc0 = kTh + (size_t)lane * T_LEN;
    const half_t* kc1 = kTh + (size_t)(32 + lane) * T_LEN;
    {
      int s0 = wave * 128;
      v16h kb0 = load_b(kc0 + s0, T_LEN), kb1 = load_b(kc1 + s0, T_LEN);
#pragma unroll
      for (int nt = 0; nt < 8; ++nt) {
        v16h nk0, nk1;
        if (nt < 7) {                            // compile-time guarded preload
          nk0 = load_b(kc0 + s0 + 16, T_LEN);
          nk1 = load_b(kc1 + s0 + 16, T_LEN);
        }
        v8f acc = {};
        acc = wmma16(a0, kb0, acc);
        acc = wmma16(a1, kb1, acc);
        int n = lane & 15, m_add = (lane >> 4) << 3;
        int s = s0 + n;
#pragma unroll
        for (int r = 0; r < 8; ++r) {
          int m = r + m_add;
          int t = t0 + m;
          int idx = s - t + 1024;                // position_x gather
          float pb =
              (idx < 1024) ? fx[idx] : ((idx == 1024) ? 0.f : bx[idx - 1025]);
          float z = ctx[t * BATCH + b] + pb + hxv;
          float sg = 1.f / (1.f + __expf(-z));
          Sld[m * 1024 + s] = acc[r] + __logf(fmaxf(sg, 1e-10f));
        }
        kb0 = nk0; kb1 = nk1;
        s0 += 16;
      }
    }
    __syncthreads();

    // ---- softmax; thread (row,sub) owns cols [sub*64, +64)
    float* srow = Sld + row * 1024;
    int c0 = sub * 64;
    float mx = -3.0e38f;
    for (int c = c0; c < c0 + 64; ++c) mx = fmaxf(mx, srow[c]);
    red[row * 16 + sub] = mx;
    __syncthreads();
    float rmx = -3.0e38f;
#pragma unroll
    for (int i = 0; i < 16; ++i) rmx = fmaxf(rmx, red[row * 16 + i]);
    __syncthreads();
    float sum = 0.f;
    for (int c = c0; c < c0 + 64; ++c) {
      float e = __expf(srow[c] - rmx);
      srow[c] = e;
      sum += e;
    }
    red[row * 16 + sub] = sum;
    __syncthreads();
    float rsum = 0.f;
#pragma unroll
    for (int i = 0; i < 16; ++i) rsum += red[row * 16 + i];
    float inv = 1.f / rsum;
    float* avgp = avg_out + ((size_t)b * T_LEN + (t0 + row)) * T_LEN;
    for (int c = c0; c < c0 + 64; ++c) {
      float p = srow[c] * inv;
      srow[c] = p;                 // keep normalized P for P@V
      float contrib = p * (1.f / HEADS);
      if (h == 0) avgp[c] = contrib; else avgp[c] += contrib;   // block owns rows
    }
    __syncthreads();

    // ---- P @ V: wave covers key range [wave*128, +128), pipelined chunks
    v8f d0 = {}, d1 = {}, d2 = {}, d3 = {};
    {
      int kc = wave * 128;
      v16h ap = pack_p(Sld, lane, kc);
      const half_t* vrow = vb + (hb + kc + lane) * HDIM;
      v16h w0 = load_b(vrow, HDIM), w1 = load_b(vrow + 16, HDIM);
      v16h w2 = load_b(vrow + 32, HDIM), w3 = load_b(vrow + 48, HDIM);
#pragma unroll
      for (int cc = 0; cc < 4; ++cc) {
        v16h apn, nw0, nw1, nw2, nw3;
        if (cc < 3) {                            // compile-time guarded preload
          apn = pack_p(Sld, lane, kc + 32);
          const half_t* vrn = vb + (hb + kc + 32 + lane) * HDIM;
          nw0 = load_b(vrn, HDIM); nw1 = load_b(vrn + 16, HDIM);
          nw2 = load_b(vrn + 32, HDIM); nw3 = load_b(vrn + 48, HDIM);
        }
        d0 = wmma16(ap, w0, d0);
        d1 = wmma16(ap, w1, d1);
        d2 = wmma16(ap, w2, d2);
        d3 = wmma16(ap, w3, d3);
        ap = apn; w0 = nw0; w1 = nw1; w2 = nw2; w3 = nw3;
        kc += 32;
      }
    }
    __syncthreads();               // all waves done reading P
    {
      int n = lane & 15, m_add = (lane >> 4) << 3;
      float* pbase = Sld + wave * 1024;   // reuse S region for partials
#pragma unroll
      for (int r = 0; r < 8; ++r) {
        int m = r + m_add;
        pbase[m * 64 + 0  + n] = d0[r];
        pbase[m * 64 + 16 + n] = d1[r];
        pbase[m * 64 + 32 + n] = d2[r];
        pbase[m * 64 + 48 + n] = d3[r];
      }
    }
    __syncthreads();
    {
      int e0 = tid * 4;            // 1024 outputs, 4 per thread
#pragma unroll
      for (int ii = 0; ii < 4; ++ii) {
        int e = e0 + ii;
        float s = 0.f;
#pragma unroll
        for (int w = 0; w < 8; ++w) s += Sld[w * 1024 + e];
        s = fmaxf(s, 0.f);         // relu before out-proj
        int m = e >> 6, d = e & 63;
        arelu[(((size_t)(t0 + m)) * BATCH + b) * EMB + h * HDIM + d] = (half_t)s;
      }
    }
    __syncthreads();
  }
}

// ------------------------------------------------------------- out-proj GEMM
// 16x64 tile per wave, double-buffered K loop; owhT is (E x E) transposed.
__global__ void __launch_bounds__(256) out_kernel(
    const half_t* __restrict__ ah, const half_t* __restrict__ owhT,
    const float* __restrict__ ob, float* __restrict__ out) {
  int lane = threadIdx.x & 31, wave = threadIdx.x >> 5;
  int gw = blockIdx.x * 8 + wave;
  int tn = gw & 15;                // EMB/64 = 16 column groups
  int tm = gw >> 4;
  int tb0 = tm * 16, o0 = tn * 64;
  v8f acc[4] = {};
  const half_t* abase = ah + (size_t)tb0 * EMB;

  v16h a = load_a(abase, EMB, lane, 0);
  const half_t* brow = owhT + (size_t)lane * EMB + o0;
  v16h b0 = load_b(brow, EMB), b1 = load_b(brow + 16, EMB);
  v16h b2 = load_b(brow + 32, EMB), b3 = load_b(brow + 48, EMB);
  for (int k0 = 0; k0 < EMB - 32; k0 += 32) {
    const half_t* brn = owhT + (size_t)(k0 + 32 + lane) * EMB + o0;
    v16h an = load_a(abase, EMB, lane, k0 + 32);
    v16h n0 = load_b(brn, EMB), n1 = load_b(brn + 16, EMB);
    v16h n2 = load_b(brn + 32, EMB), n3 = load_b(brn + 48, EMB);
    __builtin_prefetch(abase + (size_t)(lane & 15) * EMB + k0 + 64, 0, 1);
    __builtin_prefetch(brn + (size_t)32 * EMB, 0, 1);
    acc[0] = wmma16(a, b0, acc[0]);
    acc[1] = wmma16(a, b1, acc[1]);
    acc[2] = wmma16(a, b2, acc[2]);
    acc[3] = wmma16(a, b3, acc[3]);
    a = an; b0 = n0; b1 = n1; b2 = n2; b3 = n3;
  }
  acc[0] = wmma16(a, b0, acc[0]);
  acc[1] = wmma16(a, b1, acc[1]);
  acc[2] = wmma16(a, b2, acc[2]);
  acc[3] = wmma16(a, b3, acc[3]);

  int n = lane & 15, m_add = (lane >> 4) << 3;
#pragma unroll
  for (int jt = 0; jt < 4; ++jt) {
    int o = o0 + jt * 16 + n;
    float obv = ob[o];
#pragma unroll
    for (int r = 0; r < 8; ++r) {
      int tb = tb0 + r + m_add;
      out[(size_t)tb * EMB + o] = acc[jt][r] + obv;
    }
  }
}

extern "C" void kernel_launch(void* const* d_in, const int* in_sizes, int n_in,
                              void* d_out, int out_size, void* d_ws,
                              size_t ws_size, hipStream_t stream) {
  const float* query = (const float*)d_in[0];   // (T,B,E)
  const float* ipw   = (const float*)d_in[1];   // (3E,E)
  const float* ipb   = (const float*)d_in[2];   // (3E)
  const float* outw  = (const float*)d_in[3];   // (E,E)
  const float* outb  = (const float*)d_in[4];   // (E)
  const float* fx    = (const float*)d_in[5];   // (1024)
  const float* bx    = (const float*)d_in[6];   // (1024)
  const float* hx    = (const float*)d_in[7];   // (16)
  const float* cw    = (const float*)d_in[8];   // (1,E)

  char* ws = (char*)d_ws;
  const size_t MB = 1024ull * 1024ull;
  half_t* qh    = (half_t*)(ws);              // (4096,1024) f16        8 MB
  half_t* whT   = (half_t*)(ws + 8  * MB);    // (1024,3072) f16 W^T    6 MB
  half_t* qbuf  = (half_t*)(ws + 14 * MB);    // (B,H,T,D) f16          8 MB
  half_t* kTbuf = (half_t*)(ws + 22 * MB);    // (B,H,D,T) f16 K^T      8 MB
  half_t* vbuf  = (half_t*)(ws + 30 * MB);    // (B,H,T,D) f16          8 MB
  half_t* arelu = (half_t*)(ws + 38 * MB);    // (4096,1024) f16        8 MB
  half_t* owhT  = (half_t*)(ws + 46 * MB);    // (1024,1024) f16 W_o^T  2 MB
  float*  ctx   = (float*)(ws + 48 * MB);     // (4096) f32            16 KB

  float* attn_out = (float*)d_out;                           // (T,B,E)
  float* avg_out  = attn_out + (size_t)T_LEN * BATCH * EMB;  // (B,T,T)

  cvt_f32_to_f16<<<4096, 256, 0, stream>>>(query, qh, ROWS * EMB);
  transpose_f32_f16<<<(THREE_E / 32) * (EMB / 32), 256, 0, stream>>>(
      ipw, whT, THREE_E, EMB);
  transpose_f32_f16<<<(EMB / 32) * (EMB / 32), 256, 0, stream>>>(
      outw, owhT, EMB, EMB);
  ctx_kernel<<<ROWS, 256, 0, stream>>>(query, cw, ctx);
  qkv_kernel<<<(ROWS / 16) * (THREE_E / 64) / 8, 256, 0, stream>>>(
      qh, whT, ipb, qbuf, kTbuf, vbuf);
  attn_kernel<<<BATCH * (T_LEN / 16), 256, 16640 * sizeof(float), stream>>>(
      qbuf, kTbuf, vbuf, ctx, fx, bx, hx, arelu, avg_out);
  out_kernel<<<(ROWS / 16) * (EMB / 64) / 8, 256, 0, stream>>>(
      arelu, owhT, outb, attn_out);
}
